// DeepsetsHead_11098195493222
// MI455X (gfx1250) — matrix operand order repaired and checked by reading, not compiled, not run on hardware
//
#include <hip/hip_runtime.h>
#include <hip/hip_bf16.h>
#include <math.h>

// ---------------- problem constants ----------------
#define NBOX  5000          // boxes
#define MROW  5024          // rows padded to multiple of 32 (157*32) for 2x2 WMMA blocking
#define SORTN 8192          // power-of-two pad for bitonic sort
#define FEATN 1024
#define KMAX  1056          // max padded feature dim (1033 -> 1056, 33*32)

// layer dims: 1033->1000->600->300->150->1, padded to multiples of 32
static const int DIN_[5] = {1033, 1000, 600, 300, 150};
static const int DO_[5]  = {1000, 600, 300, 150, 1};
static const int KP_[5]  = {1056, 1024, 608, 320, 160};   // padded K (input dim)
static const int NP_[5]  = {1024, 608, 320, 160, 32};     // padded N (output dim); NP[l] == KP[l+1]

// ---------------- WMMA types (gfx1250, wave32) ----------------
typedef __attribute__((ext_vector_type(16))) __bf16       v16bf;
typedef __attribute__((ext_vector_type(8)))  float        v8f;
typedef __attribute__((ext_vector_type(4)))  unsigned int u32x4;

__device__ __forceinline__ unsigned short f2bf(float x) {
  __bf16 h = (__bf16)x;                       // RNE convert on VALU
  return __builtin_bit_cast(unsigned short, h);
}

// ---------------- 1) bitonic argsort of scores (descending) ----------------
__global__ void __launch_bounds__(1024)
k_sort(const float* __restrict__ cls, int* __restrict__ ord, float* __restrict__ sscore) {
  extern __shared__ char smem[];
  float* keys = (float*)smem;               // 32 KB
  int*   vals = (int*)(smem + SORTN * 4);   // 32 KB
  const int tid = threadIdx.x;
  for (int i = tid; i < SORTN; i += 1024) {
    keys[i] = (i < NBOX) ? cls[2 * i + 1] : -3.4e38f;
    vals[i] = i;
  }
  for (int size = 2; size <= SORTN; size <<= 1) {
    for (int stride = size >> 1; stride > 0; stride >>= 1) {
      __syncthreads();
      for (int t = tid; t < SORTN / 2; t += 1024) {
        int lo = t & (stride - 1);
        int i  = ((t - lo) << 1) + lo;
        int j  = i + stride;
        bool desc = ((i & size) == 0);      // final pass: all descending
        float ki = keys[i], kj = keys[j];
        if ((ki < kj) == desc) {
          keys[i] = kj; keys[j] = ki;
          int vi = vals[i]; vals[i] = vals[j]; vals[j] = vi;
        }
      }
    }
  }
  __syncthreads();
  for (int i = tid; i < NBOX; i += 1024) { ord[i] = vals[i]; sscore[i] = keys[i]; }
}

// ---------------- 2) gather sorted boxes ----------------
__global__ void k_gather(const float* __restrict__ bb, const int* __restrict__ ord,
                         float* sx1, float* sy1, float* sx2, float* sy2, float* sarea) {
  int i = blockIdx.x * blockDim.x + threadIdx.x;
  if (i >= NBOX) return;
  int o = ord[i];
  float x1 = bb[4 * o + 0], y1 = bb[4 * o + 1];
  float x2 = bb[4 * o + 2], y2 = bb[4 * o + 3];
  sx1[i] = x1; sy1[i] = y1; sx2[i] = x2; sy2[i] = y2;
  sarea[i] = (x2 - x1 + 1.f) * (y2 - y1 + 1.f);
}

// ---------------- 3) greedy IoU clustering, one WGP, all state in LDS ----------------
// LDS: 5 float arrays + 2 int arrays of MROW = 140672 B (< 320 KB/WGP)
__global__ void __launch_bounds__(1024)
k_cluster(const float* __restrict__ sx1, const float* __restrict__ sy1,
          const float* __restrict__ sx2, const float* __restrict__ sy2,
          const float* __restrict__ sarea, int* __restrict__ assign,
          float* __restrict__ counts) {
  extern __shared__ char smem[];
  float* lx1 = (float*)smem;
  float* ly1 = lx1 + MROW;
  float* lx2 = ly1 + MROW;
  float* ly2 = lx2 + MROW;
  float* lar = ly2 + MROW;
  int* lis     = (int*)(lar + MROW);
  int* lassign = lis + MROW;
  const int tid = threadIdx.x;
  for (int i = tid; i < MROW; i += 1024) {
    if (i < NBOX) {
      lx1[i] = sx1[i]; ly1[i] = sy1[i]; lx2[i] = sx2[i]; ly2[i] = sy2[i];
      lar[i] = sarea[i]; lis[i] = 1;
    } else {
      lx1[i] = 0.f; ly1[i] = 0.f; lx2[i] = -1.f; ly2[i] = -1.f; lar[i] = 0.f; lis[i] = 0;
    }
    lassign[i] = 0;
  }
  for (int j = 0; j < NBOX; ++j) {
    __syncthreads();                       // order prev writes before seed read
    int   seed = lis[j];
    float x1j = lx1[j], y1j = ly1[j], x2j = lx2[j], y2j = ly2[j], aj = lar[j];
    __syncthreads();                       // seed read before this step's writes
    if (seed) {
      for (int i = tid; i < NBOX; i += 1024) {
        float iw = fminf(x2j, lx2[i]) - fmaxf(x1j, lx1[i]) + 1.f; iw = fmaxf(iw, 0.f);
        float ih = fminf(y2j, ly2[i]) - fmaxf(y1j, ly1[i]) + 1.f; ih = fmaxf(ih, 0.f);
        float ov = iw * ih;
        float iou = ov / (aj + lar[i] - ov);
        if (iou > 0.5f) {                  // member = seed & hit & is_c ; clear seed&hit
          if (lis[i]) lassign[i] = j;
          lis[i] = 0;
        }
      }
    }
  }
  __syncthreads();
  for (int i = tid; i < NBOX; i += 1024) {
    assign[i] = lassign[i];
    atomicAdd(&counts[lassign[i]], 1.0f);
  }
}

__global__ void k_cdeninv(const float* __restrict__ counts, float* __restrict__ cden) {
  int i = blockIdx.x * blockDim.x + threadIdx.x;
  if (i >= MROW) return;
  cden[i] = 1.0f / fmaxf(counts[i], 1.0f);
}

// ---------------- 4) build feature matrix X (MROW x 1056, f32, zero-padded) ----------------
__global__ void k_build_x(const float* __restrict__ feats, const float* __restrict__ img_shape,
                          const int* __restrict__ ord,
                          const float* __restrict__ sx1, const float* __restrict__ sy1,
                          const float* __restrict__ sx2, const float* __restrict__ sy2,
                          const float* __restrict__ sscore, float* __restrict__ X) {
  long long t = (long long)blockIdx.x * blockDim.x + threadIdx.x;
  if (t >= (long long)MROW * KMAX) return;
  int row = (int)(t / KMAX), c = (int)(t % KMAX);
  float v = 0.f;
  if (row < NBOX) {
    if (c < 4) {
      v = (c == 0) ? sx1[row] : (c == 1) ? sy1[row] : (c == 2) ? sx2[row] : sy2[row];
    } else if (c < 4 + FEATN) {
      v = feats[(long long)ord[row] * FEATN + (c - 4)];
    } else if (c <= 1032) {
      float H = img_shape[0], W = img_shape[1];
      float x1 = sx1[row], y1 = sy1[row], x2 = sx2[row], y2 = sy2[row];
      float width  = (x2 / W - x1 / W) / W;
      float height = (y2 / H - y1 / H) / H;
      if      (c == 1028) v = width;
      else if (c == 1029) v = height;
      else if (c == 1030) v = width / (height + 2.2204460492503131e-16f);
      else if (c == 1031) v = width * height;
      else                v = sscore[row];
    }
  }
  X[t] = v;
}

// ---------------- 5) weight conversion to padded bf16 (B[n][k], row-major) ----------------
__global__ void k_conv_w(const float* __restrict__ Wg, const float* __restrict__ Wl,
                         unsigned short* __restrict__ Wgb, unsigned short* __restrict__ Wlb,
                         int dout, int din, int KPp, int NPp) {
  long long t = (long long)blockIdx.x * blockDim.x + threadIdx.x;
  if (t >= (long long)NPp * KPp) return;
  int n = (int)(t / KPp), k = (int)(t % KPp);
  float g = 0.f, l = 0.f;
  if (n < dout && k < din) { long long s = (long long)n * din + k; g = Wg[s]; l = Wl[s]; }
  Wgb[t] = f2bf(g);
  Wlb[t] = f2bf(l);
}

// ---------------- 6) segment sum (f32 atomics) ----------------
__global__ void k_seg(const float* __restrict__ X, const int* __restrict__ assign,
                      float* __restrict__ XM, int din, int KPp) {
  long long t = (long long)blockIdx.x * blockDim.x + threadIdx.x;
  if (t >= (long long)NBOX * din) return;
  int i = (int)(t / din), c = (int)(t % din);
  atomicAdd(&XM[(long long)assign[i] * KPp + c], X[(long long)i * KPp + c]);
}

// ---------------- 7) f32 -> bf16 activations (mean applied to XM) ----------------
__global__ void k_convert(const float* __restrict__ X, const float* __restrict__ XM,
                          const float* __restrict__ cden,
                          unsigned short* __restrict__ Xb, unsigned short* __restrict__ XMb,
                          int KPp) {
  long long t = (long long)blockIdx.x * blockDim.x + threadIdx.x;
  if (t >= (long long)MROW * KPp) return;
  int row = (int)(t / KPp);
  Xb[t]  = f2bf(X[t]);
  XMb[t] = f2bf(XM[t] * cden[row]);
}

// ---------------- 8) WMMA GEMM: C[M x N] = A[M x K](bf16) * B[N x K](bf16)^T ----------------
// 4 waves/block; each wave computes a 32x32 output block as 2x2 WMMA tiles
// (4x v_wmma_f32_16x16x32_bf16 per K-step; 16 FLOP/byte of fragment traffic).
// Fragment layouts per CDNA5 ISA 7.12.2:
//  A (16x32 bf16): lane = lo + 16*hi, row M = lo; element e = g*8+w  <-> K = g*16 + hi*8 + w
//  B (32x16 bf16): lane = lo + 16*hi, col N = lo; element e = g*8+w <-> K = g*16 + hi*8 + w
//  C (16x16 f32):  lane = lo + 16*hi, col N = lo; element r          <-> M = r + 8*hi
__global__ void __launch_bounds__(128)
k_gemm(const unsigned short* __restrict__ A, const unsigned short* __restrict__ B,
       float* __restrict__ C, int K, int Nn) {
  const int wave = threadIdx.x >> 5;
  const int lane = threadIdx.x & 31;
  const int lo = lane & 15, hi = lane >> 4;
  const int colBase = blockIdx.x * 32;
  const int rowBase = (blockIdx.y * 4 + wave) * 32;
  if (rowBase >= MROW) return;               // uniform per wave: EXEC stays all-1s for WMMA

  const unsigned short* A0 = A + (long long)(rowBase + lo) * K;
  const unsigned short* A1 = A0 + (long long)16 * K;
  const unsigned short* B0 = B + (long long)(colBase + lo) * K;
  const unsigned short* B1 = B0 + (long long)16 * K;

  v8f c00 = {}, c01 = {}, c10 = {}, c11 = {};
  union UA { v16bf v; u32x4 q[2]; };
  for (int k0 = 0; k0 < K; k0 += 32) {
    UA a0, a1, b0, b1;
    a0.q[0] = *(const u32x4*)(A0 + k0 + hi * 8);        // K = g*16 + hi*8 + w, g=0
    a0.q[1] = *(const u32x4*)(A0 + k0 + 16 + hi * 8);   // g=1
    a1.q[0] = *(const u32x4*)(A1 + k0 + hi * 8);
    a1.q[1] = *(const u32x4*)(A1 + k0 + 16 + hi * 8);
    b0.q[0] = *(const u32x4*)(B0 + k0 + hi * 8);        // K = g*16 + hi*8 + w, g=0
    b0.q[1] = *(const u32x4*)(B0 + k0 + 16 + hi * 8);   // g=1
    b1.q[0] = *(const u32x4*)(B1 + k0 + hi * 8);
    b1.q[1] = *(const u32x4*)(B1 + k0 + 16 + hi * 8);
    if (k0 + 32 < K) {                                  // gfx1250 global_prefetch_b8 (near-cache)
      __builtin_prefetch(A0 + k0 + 96, 0, 3);
      __builtin_prefetch(A1 + k0 + 96, 0, 3);
      __builtin_prefetch(B0 + k0 + 96, 0, 3);
      __builtin_prefetch(B1 + k0 + 96, 0, 3);
    }
    c00 = __builtin_amdgcn_wmma_f32_16x16x32_bf16(false, a0.v, false, b0.v, (short)0, c00, false, false);
    c01 = __builtin_amdgcn_wmma_f32_16x16x32_bf16(false, a0.v, false, b1.v, (short)0, c01, false, false);
    c10 = __builtin_amdgcn_wmma_f32_16x16x32_bf16(false, a1.v, false, b0.v, (short)0, c10, false, false);
    c11 = __builtin_amdgcn_wmma_f32_16x16x32_bf16(false, a1.v, false, b1.v, (short)0, c11, false, false);
  }
  float* Crow = C + (long long)rowBase * Nn + colBase;
#pragma unroll
  for (int r = 0; r < 8; ++r) {
    int m0 = r + 8 * hi;
    Crow[(long long)m0 * Nn + lo]             = c00[r];
    Crow[(long long)m0 * Nn + 16 + lo]        = c01[r];
    Crow[(long long)(m0 + 16) * Nn + lo]      = c10[r];
    Crow[(long long)(m0 + 16) * Nn + 16 + lo] = c11[r];
  }
}

// ---------------- 9) combine: Z = elu(Z + bg - S[assign]) with padding zeroed ----------------
__global__ void k_combine(float* __restrict__ Z, const float* __restrict__ S,
                          const float* __restrict__ bg, const int* __restrict__ assign,
                          int dout, int NPp) {
  long long t = (long long)blockIdx.x * blockDim.x + threadIdx.x;
  if (t >= (long long)MROW * NPp) return;
  int row = (int)(t / NPp), n = (int)(t % NPp);
  float v = 0.f;
  if (row < NBOX && n < dout) {
    v = Z[t] + bg[n] - S[(long long)assign[row] * NPp + n];
    v = (v > 0.f) ? v : expm1f(v);           // ELU, alpha=1
  }
  Z[t] = v;
}

// ---------------- 10) final mask by cluster count ----------------
__global__ void k_output(const float* __restrict__ Xf, const float* __restrict__ counts,
                         const int* __restrict__ assign, float* __restrict__ out) {
  int i = blockIdx.x * blockDim.x + threadIdx.x;
  if (i >= NBOX) return;
  out[i] = (counts[assign[i]] >= 2.0f) ? Xf[(long long)i * 32] : 0.f;
}

// ---------------- host ----------------
extern "C" void kernel_launch(void* const* d_in, const int* in_sizes, int n_in,
                              void* d_out, int out_size, void* d_ws, size_t ws_size,
                              hipStream_t stream) {
  (void)in_sizes; (void)n_in; (void)out_size; (void)ws_size;

  const float* bboxes = (const float*)d_in[0];
  const float* cls    = (const float*)d_in[1];
  const float* feats  = (const float*)d_in[2];
  const float* imgsh  = (const float*)d_in[3];
  const float* Wg[5]; const float* bg[5]; const float* Wl[5];
  for (int l = 0; l < 5; ++l) {
    Wg[l] = (const float*)d_in[4 + 3 * l];
    bg[l] = (const float*)d_in[5 + 3 * l];
    Wl[l] = (const float*)d_in[6 + 3 * l];
  }

  char* ws = (char*)d_ws;
  size_t o = 0;
  auto nxt = [&](size_t bytes) { size_t r = o; o += (bytes + 255) & ~(size_t)255; return r; };

  int*   ord    = (int*)  (ws + nxt((size_t)MROW * 4));
  float* sscore = (float*)(ws + nxt((size_t)MROW * 4));
  float* sx1    = (float*)(ws + nxt((size_t)MROW * 4));
  float* sy1    = (float*)(ws + nxt((size_t)MROW * 4));
  float* sx2    = (float*)(ws + nxt((size_t)MROW * 4));
  float* sy2    = (float*)(ws + nxt((size_t)MROW * 4));
  float* sarea  = (float*)(ws + nxt((size_t)MROW * 4));
  int*   assign = (int*)  (ws + nxt((size_t)MROW * 4));
  float* counts = (float*)(ws + nxt((size_t)MROW * 4));
  float* cden   = (float*)(ws + nxt((size_t)MROW * 4));
  float* X0     = (float*)(ws + nxt((size_t)MROW * KMAX * 4));
  float* X1     = (float*)(ws + nxt((size_t)MROW * KMAX * 4));
  float* XM     = (float*)(ws + nxt((size_t)MROW * KMAX * 4));
  unsigned short* Xb  = (unsigned short*)(ws + nxt((size_t)MROW * KMAX * 2));
  unsigned short* XMb = (unsigned short*)(ws + nxt((size_t)MROW * KMAX * 2));
  float* S      = (float*)(ws + nxt((size_t)MROW * 1024 * 4));
  unsigned short* Wgb = (unsigned short*)(ws + nxt((size_t)1024 * KMAX * 2));
  unsigned short* Wlb = (unsigned short*)(ws + nxt((size_t)1024 * KMAX * 2));

  // sort + gather + cluster
  k_sort<<<1, 1024, SORTN * 8, stream>>>(cls, ord, sscore);
  k_gather<<<(NBOX + 255) / 256, 256, 0, stream>>>(bboxes, ord, sx1, sy1, sx2, sy2, sarea);
  hipMemsetAsync(counts, 0, (size_t)MROW * 4, stream);
  k_cluster<<<1, 1024, (size_t)MROW * 4 * 7, stream>>>(sx1, sy1, sx2, sy2, sarea, assign, counts);
  k_cdeninv<<<(MROW + 255) / 256, 256, 0, stream>>>(counts, cden);

  // build X (layer 0 input)
  {
    long long tot = (long long)MROW * KMAX;
    k_build_x<<<(unsigned)((tot + 255) / 256), 256, 0, stream>>>(
        feats, imgsh, ord, sx1, sy1, sx2, sy2, sscore, X0);
  }

  // 5 perm-equivariant layers
  float* Xcur = X0;
  float* Xnxt = X1;
  for (int l = 0; l < 5; ++l) {
    const int din = DIN_[l], dout = DO_[l], kp = KP_[l], np = NP_[l];

    long long wtot = (long long)np * kp;
    k_conv_w<<<(unsigned)((wtot + 255) / 256), 256, 0, stream>>>(Wg[l], Wl[l], Wgb, Wlb, dout, din, kp, np);

    hipMemsetAsync(XM, 0, (size_t)MROW * kp * 4, stream);
    long long stot = (long long)NBOX * din;
    k_seg<<<(unsigned)((stot + 255) / 256), 256, 0, stream>>>(Xcur, assign, XM, din, kp);

    long long ctot = (long long)MROW * kp;
    k_convert<<<(unsigned)((ctot + 255) / 256), 256, 0, stream>>>(Xcur, XM, cden, Xb, XMb, kp);

    dim3 gg((unsigned)(np / 32), (unsigned)((MROW / 32 + 3) / 4));
    k_gemm<<<gg, 128, 0, stream>>>(Xb, Wgb, Xnxt, kp, np);   // Z  = x  @ Wg^T
    k_gemm<<<gg, 128, 0, stream>>>(XMb, Wlb, S, kp, np);     // S  = xm @ Wl^T

    long long btot = (long long)MROW * np;
    k_combine<<<(unsigned)((btot + 255) / 256), 256, 0, stream>>>(Xnxt, S, bg[l], assign, dout, np);

    float* tmp = Xcur; Xcur = Xnxt; Xnxt = tmp;
  }

  k_output<<<(NBOX + 255) / 256, 256, 0, stream>>>(Xcur, counts, assign, (float*)d_out);
}
